// Decoder_59072980189993
// MI455X (gfx1250) — matrix operand order repaired and checked by reading, not compiled, not run on hardware
//
#include <hip/hip_runtime.h>
#include <hip/hip_bf16.h>
#include <math.h>

typedef __attribute__((ext_vector_type(16))) _Float16 v16h;
typedef __attribute__((ext_vector_type(8)))  float    v8f;

// ---------------- problem constants ----------------
constexpr int B     = 32;
constexpr int TENC  = 512;
constexpr int TS    = 501;          // scan steps (T_DEC + 1)
constexpr int NMEL  = 160;
constexpr int HID   = 1024;
constexpr int G4    = 4096;
constexpr int KA    = 1792;         // 256 (prenet) + 512 (ctx) + 1024 (h)
constexpr int KD    = 2560;         // 512 (ctx) + 1024 (ah) + 1024 (dh)
constexpr int KM    = 1536;         // 1024 (dh) + 512 (ctx)
constexpr int MROWS = TS * B;       // 16032 prenet rows
constexpr int LDS_BYTES = B * KD * 2;   // 160 KB dynamic LDS (max of xa/xd)

constexpr size_t alup(size_t x){ return (x + 255) & ~(size_t)255; }

// ---------------- workspace layout ----------------
constexpr size_t OFF_WA  = 0;
constexpr size_t OFF_WD  = alup(OFF_WA  + (size_t)KA*G4*2);
constexpr size_t OFF_WQ  = alup(OFF_WD  + (size_t)KD*G4*2);
constexpr size_t OFF_WK  = alup(OFF_WQ  + (size_t)1024*128*2);
constexpr size_t OFF_WM  = alup(OFF_WK  + (size_t)512*128*2);
constexpr size_t OFF_WP1 = alup(OFF_WM  + (size_t)KM*160*2);
constexpr size_t OFF_WP2 = alup(OFF_WP1 + (size_t)160*256*2);
constexpr size_t OFF_H1  = alup(OFF_WP2 + (size_t)256*256*2);
constexpr size_t OFF_PRE = alup(OFF_H1  + (size_t)MROWS*256*2);
constexpr size_t OFF_MEMK= alup(OFF_PRE + (size_t)MROWS*256*2);
constexpr size_t OFF_BA  = alup(OFF_MEMK+ (size_t)B*TENC*128*4);
constexpr size_t OFF_BD  = alup(OFF_BA  + (size_t)G4*4);
constexpr size_t OFF_CTX = alup(OFF_BD  + (size_t)G4*4);
constexpr size_t OFF_AH  = alup(OFF_CTX + (size_t)B*512*4);
constexpr size_t OFF_AC  = alup(OFF_AH  + (size_t)B*HID*4);
constexpr size_t OFF_DH  = alup(OFF_AC  + (size_t)B*HID*4);
constexpr size_t OFF_DC  = alup(OFF_DH  + (size_t)B*HID*4);
constexpr size_t OFF_AWV = alup(OFF_DC  + (size_t)B*HID*4);
constexpr size_t OFF_AWT = alup(OFF_AWV + (size_t)B*TENC*4);
constexpr size_t OFF_G   = alup(OFF_AWT + (size_t)B*TENC*4);
constexpr size_t OFF_XA  = alup(OFF_G   + (size_t)B*G4*4);
constexpr size_t OFF_XD  = alup(OFF_XA  + (size_t)B*KA*2);
constexpr size_t OFF_AHF = alup(OFF_XD  + (size_t)B*KD*2);
constexpr size_t OFF_HF  = alup(OFF_AHF + (size_t)B*HID*2);
constexpr size_t OFF_Q   = alup(OFF_HF  + (size_t)B*KM*2);
constexpr size_t OFF_LC  = alup(OFF_Q   + (size_t)B*128*4);
constexpr size_t OFF_E   = alup(OFF_LC  + (size_t)B*TENC*32*4);
constexpr size_t OFF_BAR = alup(OFF_E   + (size_t)B*TENC*4);

// ---------------- WMMA helpers ----------------
// A-fragment (16x32 f16): lane (m = lane&15, half = lane>>4) holds
// k = k0 + half*8 + i (i<8) and k = k0 + 16 + half*8 + (i-8) (i>=8).
__device__ inline v16h a_frag_f16(const _Float16* row, int k0, int half){
  v16h a;
  const _Float16* p = row + k0 + half*8;
  #pragma unroll
  for (int i = 0; i < 8; ++i){ a[i] = p[i]; a[i+8] = p[16+i]; }
  return a;
}
__device__ inline v16h a_frag_f32(const float* row, int k0, int half){
  v16h a;
  if (row == nullptr){
    #pragma unroll
    for (int i = 0; i < 16; ++i) a[i] = (_Float16)0.0f;
    return a;
  }
  const float* p = row + k0 + half*8;
  #pragma unroll
  for (int i = 0; i < 8; ++i){ a[i] = (_Float16)p[i]; a[i+8] = (_Float16)p[16+i]; }
  return a;
}
// B-fragments are pre-packed: frag for (kt,nt) is 32 lanes x 16 f16 contiguous.
__device__ inline v16h b_frag(const _Float16* Wp, int NT, int kt, int nt, int lane){
  return *reinterpret_cast<const v16h*>(Wp + (((size_t)kt*NT + nt)*32 + lane)*16);
}
__device__ inline v8f wmma_f16(v16h a, v16h b, v8f c){
  return __builtin_amdgcn_wmma_f32_16x16x32_f16(false, a, false, b, (short)0, c, false, false);
}
__device__ inline float sigm(float x){ return 1.0f / (1.0f + __expf(-x)); }

// ---------------- grid-wide software barrier ----------------
__device__ inline void gsync(unsigned* bar){
  __syncthreads();
  if (threadIdx.x == 0){
    volatile unsigned* vgen = bar + 1;
    unsigned g = *vgen;
    __threadfence();
    if (atomicAdd(bar, 1u) == gridDim.x - 1){
      *bar = 0;
      __threadfence();
      atomicAdd(bar + 1, 1u);
    } else {
      while (*vgen == g) { __builtin_amdgcn_s_sleep(1); }
    }
  }
  __syncthreads();
}

// ---------------- prep: pack weight (N,K) row-major -> f16 WMMA-B frags ----
// Fused [W1 | W2] along K (W2 used for k >= K1).
__global__ __launch_bounds__(256)
void pack_kernel(const float* __restrict__ W1, const float* __restrict__ W2,
                 int K1, int K, int N, _Float16* __restrict__ out){
  size_t total = (size_t)K * N;
  int NT = N >> 4;
  for (size_t idx = (size_t)blockIdx.x*blockDim.x + threadIdx.x; idx < total;
       idx += (size_t)gridDim.x*blockDim.x){
    int i  = (int)(idx & 15);
    int l  = (int)((idx >> 4) & 31);
    size_t rest = idx >> 9;
    int nt = (int)(rest % NT);
    int kt = (int)(rest / NT);
    int n  = nt*16 + (l & 15);
    int k  = kt*32 + (l >> 4)*8 + (i & 7) + (i >> 3)*16;
    float v = (k < K1) ? W1[(size_t)n*K1 + k] : W2[(size_t)n*(K - K1) + (k - K1)];
    out[idx] = (_Float16)v;
  }
}

// ---------------- prep: state zero + fused biases ----------------
__global__ __launch_bounds__(256)
void init_kernel(char* ws, const float* biha, const float* bhha,
                 const float* bihd, const float* bhhd){
  float* bacomb = (float*)(ws + OFF_BA);
  float* bdcomb = (float*)(ws + OFF_BD);
  float* ctx = (float*)(ws + OFF_CTX);
  float* ah  = (float*)(ws + OFF_AH);
  float* ac  = (float*)(ws + OFF_AC);
  float* dh  = (float*)(ws + OFF_DH);
  float* dc  = (float*)(ws + OFF_DC);
  float* aw  = (float*)(ws + OFF_AWV);
  float* awt = (float*)(ws + OFF_AWT);
  unsigned* bar = (unsigned*)(ws + OFF_BAR);
  int gt = blockIdx.x*blockDim.x + threadIdx.x;
  int st = gridDim.x*blockDim.x;
  for (int i = gt; i < G4; i += st){ bacomb[i] = biha[i] + bhha[i]; bdcomb[i] = bihd[i] + bhhd[i]; }
  for (int i = gt; i < B*HID; i += st){ ah[i]=0.f; ac[i]=0.f; dh[i]=0.f; dc[i]=0.f; }
  for (int i = gt; i < B*TENC; i += st){ ctx[i]=0.f; aw[i]=0.f; awt[i]=0.f; }
  if (gt < 8) bar[gt] = 0u;
}

// ---------------- prenet layer 1: relu(mel_seq @ Wp1^T + b1) -> f16 -------
__global__ __launch_bounds__(256)
void prenet1_kernel(const float* __restrict__ mels, const _Float16* __restrict__ Wp,
                    const float* __restrict__ b1, _Float16* __restrict__ h1){
  int lane = threadIdx.x & 31;
  int wave = (blockIdx.x*blockDim.x + threadIdx.x) >> 5;
  int nwaves = (gridDim.x*blockDim.x) >> 5;
  const int MT = MROWS/16, NT = 16, KT = 5;         // K=160, N=256
  int half = lane >> 4;
  for (int task = wave; task < MT*NT; task += nwaves){
    int mt = task / NT, nt = task % NT;
    int m0 = mt*16;
    int rr = m0 + (lane & 15);
    int tt = rr >> 5, bb = rr & 31;                 // row = t*32 + b
    const float* row = (tt == 0) ? nullptr
                       : mels + ((size_t)bb*500 + (tt-1))*NMEL;
    v8f acc = {0.f,0.f,0.f,0.f,0.f,0.f,0.f,0.f};
    for (int kt = 0; kt < KT; ++kt)
      acc = wmma_f16(a_frag_f32(row, kt*32, half), b_frag(Wp, NT, kt, nt, lane), acc);
    int n = nt*16 + (lane & 15);
    #pragma unroll
    for (int r = 0; r < 8; ++r){
      int m = m0 + half*8 + r;
      h1[(size_t)m*256 + n] = (_Float16)fmaxf(acc[r] + b1[n], 0.0f);
    }
  }
}

// ---------------- prenet layer 2: relu(h1 @ Wp2^T + b2) -> f16 ------------
__global__ __launch_bounds__(256)
void prenet2_kernel(const _Float16* __restrict__ h1, const _Float16* __restrict__ Wp,
                    const float* __restrict__ b2, _Float16* __restrict__ pre){
  int lane = threadIdx.x & 31;
  int wave = (blockIdx.x*blockDim.x + threadIdx.x) >> 5;
  int nwaves = (gridDim.x*blockDim.x) >> 5;
  const int MT = MROWS/16, NT = 16, KT = 8;         // K=256, N=256
  int half = lane >> 4;
  for (int task = wave; task < MT*NT; task += nwaves){
    int mt = task / NT, nt = task % NT;
    int m0 = mt*16;
    const _Float16* row = h1 + (size_t)(m0 + (lane & 15))*256;
    v8f acc = {0.f,0.f,0.f,0.f,0.f,0.f,0.f,0.f};
    for (int kt = 0; kt < KT; ++kt)
      acc = wmma_f16(a_frag_f16(row, kt*32, half), b_frag(Wp, NT, kt, nt, lane), acc);
    int n = nt*16 + (lane & 15);
    #pragma unroll
    for (int r = 0; r < 8; ++r){
      int m = m0 + half*8 + r;
      pre[(size_t)m*256 + n] = (_Float16)fmaxf(acc[r] + b2[n], 0.0f);
    }
  }
}

// ---------------- mem_keys = enc @ wk^T + bk (16384 x 128) ----------------
__global__ __launch_bounds__(256)
void memkeys_kernel(const float* __restrict__ enc, const _Float16* __restrict__ Wp,
                    const float* __restrict__ bk, float* __restrict__ memk){
  int lane = threadIdx.x & 31;
  int wave = (blockIdx.x*blockDim.x + threadIdx.x) >> 5;
  int nwaves = (gridDim.x*blockDim.x) >> 5;
  const int MT = (B*TENC)/16, NT = 8, KT = 16;      // K=512, N=128
  int half = lane >> 4;
  for (int task = wave; task < MT*NT; task += nwaves){
    int mt = task >> 3, nt = task & 7;
    int m0 = mt*16;
    const float* row = enc + (size_t)(m0 + (lane & 15))*512;
    v8f acc = {0.f,0.f,0.f,0.f,0.f,0.f,0.f,0.f};
    for (int kt = 0; kt < KT; ++kt)
      acc = wmma_f16(a_frag_f32(row, kt*32, half), b_frag(Wp, NT, kt, nt, lane), acc);
    int n = nt*16 + (lane & 15);
    #pragma unroll
    for (int r = 0; r < 8; ++r)
      memk[(size_t)(m0 + half*8 + r)*128 + n] = acc[r] + bk[n];
  }
}

// ------- big gate GEMM from LDS-staged A: [32 x K] x [K x 4096] -----------
// 512 wave-tasks: nt = wave>>1 (256 column tiles), m-tile = wave&1.
// A comes from LDS (staged once per block); B fragments stream from L2 with
// a one-ktile-ahead global_prefetch_b8.
__device__ inline void gate_gemm_lds(const _Float16* Xl, int ldx,
                                     const _Float16* Wp, int KT,
                                     const float* bias, float* G,
                                     int wave, int lane){
  int nt = wave >> 1;
  int m0 = (wave & 1) * 16;
  int half = lane >> 4;
  const _Float16* row = Xl + (size_t)(m0 + (lane & 15))*ldx;
  v8f acc = {0.f,0.f,0.f,0.f,0.f,0.f,0.f,0.f};
  for (int kt = 0; kt < KT; ++kt){
    const _Float16* bf = Wp + (((size_t)kt*256 + nt)*32 + lane)*16;
    if (kt + 1 < KT)
      __builtin_prefetch((const void*)(bf + (size_t)256*32*16), 0, 0);
    v16h b = *reinterpret_cast<const v16h*>(bf);
    acc = wmma_f16(a_frag_f16(row, kt*32, half), b, acc);
  }
  int n = nt*16 + (lane & 15);
  #pragma unroll
  for (int r = 0; r < 8; ++r)
    G[(size_t)(m0 + half*8 + r)*G4 + n] = acc[r] + bias[n];
}

// ---------------- persistent sequential decoder ----------------
__global__ __launch_bounds__(256)
void decoder_kernel(char* ws, const float* __restrict__ enc,
                    const unsigned char* __restrict__ mask,
                    const float* __restrict__ bq,
                    const float* __restrict__ convw, const float* __restrict__ convb,
                    const float* __restrict__ wl, const float* __restrict__ bl,
                    const float* __restrict__ wa, const float* __restrict__ ba,
                    const float* __restrict__ bm, const float* __restrict__ wg,
                    const float* __restrict__ bg,
                    float* __restrict__ out_m, float* __restrict__ out_g,
                    float* __restrict__ out_a){
  const _Float16* WAp = (const _Float16*)(ws + OFF_WA);
  const _Float16* WDp = (const _Float16*)(ws + OFF_WD);
  const _Float16* WQp = (const _Float16*)(ws + OFF_WQ);
  const _Float16* WMp = (const _Float16*)(ws + OFF_WM);
  const _Float16* pre = (const _Float16*)(ws + OFF_PRE);
  const float* memk   = (const float*)(ws + OFF_MEMK);
  const float* bacomb = (const float*)(ws + OFF_BA);
  const float* bdcomb = (const float*)(ws + OFF_BD);
  float* ctx = (float*)(ws + OFF_CTX);
  float* ah  = (float*)(ws + OFF_AH);
  float* ac  = (float*)(ws + OFF_AC);
  float* dh  = (float*)(ws + OFF_DH);
  float* dc  = (float*)(ws + OFF_DC);
  float* aw  = (float*)(ws + OFF_AWV);
  float* awt = (float*)(ws + OFF_AWT);
  float* G   = (float*)(ws + OFF_G);
  _Float16* xa  = (_Float16*)(ws + OFF_XA);
  _Float16* xd  = (_Float16*)(ws + OFF_XD);
  _Float16* ahf = (_Float16*)(ws + OFF_AHF);
  _Float16* hf  = (_Float16*)(ws + OFF_HF);
  float* q   = (float*)(ws + OFF_Q);
  float* lc  = (float*)(ws + OFF_LC);
  float* e   = (float*)(ws + OFF_E);
  unsigned* bar = (unsigned*)(ws + OFF_BAR);

  const int tid  = threadIdx.x;
  const int gtid = blockIdx.x*256 + tid;
  const int wave = gtid >> 5;           // 0..511
  const int lane = tid & 31;
  const int half = lane >> 4;
  __shared__ float red[256];
  extern __shared__ char smem[];        // LDS_BYTES dynamic LDS (A staging)
  _Float16* Alds = (_Float16*)smem;

  for (int t = 0; t < TS; ++t){
    // P1: build xa = f16([pre_t | ctx | ah])  (32 x 1792)
    for (int idx = gtid; idx < B*KA; idx += 16384){
      int b = idx / KA, k = idx % KA;
      float v;
      if (k < 256)      v = (float)pre[((size_t)t*B + b)*256 + k];
      else if (k < 768) v = ctx[b*512 + (k - 256)];
      else              v = ah[b*HID + (k - 768)];
      xa[idx] = (_Float16)v;
    }
    gsync(bar);

    // P2: stage xa into LDS once per block, then attention-LSTM gate GEMM
    {
      const uint4* s = (const uint4*)xa;
      uint4* d = (uint4*)Alds;
      for (int i = tid; i < (B*KA*2)/16; i += 256) d[i] = s[i];
    }
    __syncthreads();
    gate_gemm_lds(Alds, KA, WAp, KA/32, bacomb, G, wave, lane);
    gsync(bar);

    // P3: attention-LSTM pointwise
    for (int idx = gtid; idx < B*HID; idx += 16384){
      int b = idx >> 10, j = idx & 1023;
      const float* gb = G + (size_t)b*G4;
      float ig = sigm(gb[j]), fg = sigm(gb[1024 + j]);
      float gg = tanhf(gb[2048 + j]), og = sigm(gb[3072 + j]);
      float c = fg*ac[idx] + ig*gg;
      float h = og*tanhf(c);
      ac[idx] = c; ah[idx] = h; ahf[idx] = (_Float16)h;
    }
    gsync(bar);

    // P4: query GEMM (waves 0..15) + location conv (all threads)
    if (wave < 16){
      int nt = wave >> 1, m0 = (wave & 1)*16;
      const _Float16* row = ahf + (size_t)(m0 + (lane & 15))*HID;
      v8f acc = {0.f,0.f,0.f,0.f,0.f,0.f,0.f,0.f};
      for (int kt = 0; kt < 32; ++kt)
        acc = wmma_f16(a_frag_f16(row, kt*32, half), b_frag(WQp, 8, kt, nt, lane), acc);
      int n = nt*16 + (lane & 15);
      #pragma unroll
      for (int r = 0; r < 8; ++r)
        q[(size_t)(m0 + half*8 + r)*128 + n] = acc[r] + bq[n];
    }
    for (int idx = gtid; idx < B*TENC*32; idx += 16384){
      int ch = idx & 31, te = (idx >> 5) & 511, b = idx >> 14;
      float s = convb[ch];
      const float* c0 = convw + ch*62;      // conv_w (32,2,31)
      const float* awb  = aw  + b*TENC;
      const float* awtb = awt + b*TENC;
      #pragma unroll
      for (int dk = 0; dk < 31; ++dk){
        int p = te + dk - 15;
        if (p >= 0 && p < TENC) s += awb[p]*c0[dk] + awtb[p]*c0[31 + dk];
      }
      lc[idx] = s;
    }
    gsync(bar);

    // P5: energies  e[b,te] = wa . tanh(q + mem_keys + wl@lc + bl) + ba
    {
      int idx = gtid;                      // exactly 16384 items
      int b = idx >> 9;
      float lcl[32];
      const float* lcp = lc + (size_t)idx*32;
      #pragma unroll
      for (int c2 = 0; c2 < 32; ++c2) lcl[c2] = lcp[c2];
      const float* qb = q + b*128;
      const float* mk = memk + (size_t)idx*128;
      float ev = ba[0];
      for (int d = 0; d < 128; ++d){
        float v = qb[d] + mk[d] + bl[d];
        const float* wld = wl + d*32;
        #pragma unroll
        for (int c2 = 0; c2 < 32; ++c2) v += lcl[c2]*wld[c2];
        ev += wa[d]*tanhf(v);
      }
      if (mask[idx]) ev = -INFINITY;
      e[idx] = ev;
    }
    gsync(bar);

    // P6: softmax rows (block b handles batch row b), update awt, emit attn
    if (blockIdx.x < B){
      int b = blockIdx.x;
      float m1 = -INFINITY;
      for (int i = tid; i < TENC; i += 256) m1 = fmaxf(m1, e[b*TENC + i]);
      red[tid] = m1; __syncthreads();
      for (int s = 128; s > 0; s >>= 1){ if (tid < s) red[tid] = fmaxf(red[tid], red[tid+s]); __syncthreads(); }
      float mx = red[0]; __syncthreads();
      float sum = 0.f;
      for (int i = tid; i < TENC; i += 256){
        float v = __expf(e[b*TENC + i] - mx);
        e[b*TENC + i] = v; sum += v;
      }
      red[tid] = sum; __syncthreads();
      for (int s = 128; s > 0; s >>= 1){ if (tid < s) red[tid] += red[tid+s]; __syncthreads(); }
      float inv = 1.0f / red[0]; __syncthreads();
      for (int i = tid; i < TENC; i += 256){
        float w = e[b*TENC + i]*inv;
        aw[b*TENC + i] = w;
        awt[b*TENC + i] += w;
        out_a[((size_t)t*B + b)*TENC + i] = w;
      }
    }
    gsync(bar);

    // P7: context einsum + build xd = f16([ctx | ah | dh]), hf ctx half
    {
      int idx = gtid;                      // 16384 = (b,d)
      int b = idx >> 9, d = idx & 511;
      const float* awb = aw + b*TENC;
      float s = 0.f;
      for (int te = 0; te < TENC; ++te)
        s += awb[te] * enc[((size_t)b*TENC + te)*512 + d];
      ctx[idx] = s;
      xd[(size_t)b*KD + d] = (_Float16)s;
      hf[(size_t)b*KM + HID + d] = (_Float16)s;
    }
    for (int idx = gtid; idx < B*2048; idx += 16384){
      int b = idx >> 11, j = idx & 2047;
      if (j < 1024) xd[(size_t)b*KD + 512  + j]        = (_Float16)ah[b*HID + j];
      else          xd[(size_t)b*KD + 1536 + (j-1024)] = (_Float16)dh[b*HID + (j-1024)];
    }
    gsync(bar);

    // P8: stage xd into LDS once per block, then decoder-LSTM gate GEMM
    {
      const uint4* s = (const uint4*)xd;
      uint4* d = (uint4*)Alds;
      for (int i = tid; i < (B*KD*2)/16; i += 256) d[i] = s[i];
    }
    __syncthreads();
    gate_gemm_lds(Alds, KD, WDp, KD/32, bdcomb, G, wave, lane);
    gsync(bar);

    // P9: decoder-LSTM pointwise, hf dh half
    for (int idx = gtid; idx < B*HID; idx += 16384){
      int b = idx >> 10, j = idx & 1023;
      const float* gb = G + (size_t)b*G4;
      float ig = sigm(gb[j]), fg = sigm(gb[1024 + j]);
      float gg = tanhf(gb[2048 + j]), og = sigm(gb[3072 + j]);
      float c = fg*dc[idx] + ig*gg;
      float h = og*tanhf(c);
      dc[idx] = c; dh[idx] = h;
      hf[(size_t)b*KM + j] = (_Float16)h;
    }
    gsync(bar);

    // P10: mel projection (WMMA, waves 0..19) + gate scalar (waves 20..51)
    if (wave < 20){
      int nt = wave >> 1, m0 = (wave & 1)*16;   // N=160 -> 10 ntiles
      const _Float16* row = hf + (size_t)(m0 + (lane & 15))*KM;
      v8f acc = {0.f,0.f,0.f,0.f,0.f,0.f,0.f,0.f};
      for (int kt = 0; kt < KM/32; ++kt)
        acc = wmma_f16(a_frag_f16(row, kt*32, half), b_frag(WMp, 10, kt, nt, lane), acc);
      int n = nt*16 + (lane & 15);
      #pragma unroll
      for (int r = 0; r < 8; ++r)
        out_m[((size_t)t*B + m0 + half*8 + r)*NMEL + n] = acc[r] + bm[n];
    } else if (wave < 52){
      int b = wave - 20;
      const _Float16* hb = hf + (size_t)b*KM;
      float s = 0.f;
      for (int k2 = lane; k2 < KM; k2 += 32) s += wg[k2]*(float)hb[k2];
      #pragma unroll
      for (int off = 16; off; off >>= 1) s += __shfl_xor(s, off, 32);
      if (lane == 0) out_g[(size_t)t*B + b] = s + bg[0];
    }
    gsync(bar);
  }
}

// ---------------- host launcher ----------------
extern "C" void kernel_launch(void* const* d_in, const int* in_sizes, int n_in,
                              void* d_out, int out_size, void* d_ws, size_t ws_size,
                              hipStream_t stream){
  const float* enc    = (const float*)d_in[0];
  const float* mels   = (const float*)d_in[1];
  const unsigned char* mask = (const unsigned char*)d_in[2];
  const float* w_pre1 = (const float*)d_in[3];
  const float* b_pre1 = (const float*)d_in[4];
  const float* w_pre2 = (const float*)d_in[5];
  const float* b_pre2 = (const float*)d_in[6];
  const float* wih_a  = (const float*)d_in[7];
  const float* whh_a  = (const float*)d_in[8];
  const float* bih_a  = (const float*)d_in[9];
  const float* bhh_a  = (const float*)d_in[10];
  const float* wq     = (const float*)d_in[11];
  const float* bq     = (const float*)d_in[12];
  const float* wk     = (const float*)d_in[13];
  const float* bk     = (const float*)d_in[14];
  const float* convw  = (const float*)d_in[15];
  const float* convb  = (const float*)d_in[16];
  const float* wl     = (const float*)d_in[17];
  const float* bl     = (const float*)d_in[18];
  const float* wa     = (const float*)d_in[19];
  const float* ba     = (const float*)d_in[20];
  const float* wih_d  = (const float*)d_in[21];
  const float* whh_d  = (const float*)d_in[22];
  const float* bih_d  = (const float*)d_in[23];
  const float* bhh_d  = (const float*)d_in[24];
  const float* wm     = (const float*)d_in[25];
  const float* bm     = (const float*)d_in[26];
  const float* wg     = (const float*)d_in[27];
  const float* bg     = (const float*)d_in[28];

  char* ws = (char*)d_ws;
  float* out_m = (float*)d_out;                     // (501,32,160)
  float* out_g = out_m + (size_t)TS*B*NMEL;         // (501,32,1)
  float* out_a = out_g + (size_t)TS*B;              // (501,32,512)

  // Pack all GEMM weights into f16 WMMA-B fragment order (L2-resident after).
  pack_kernel<<<2048, 256, 0, stream>>>(wih_a, whh_a,  768, KA,   G4,  (_Float16*)(ws + OFF_WA));
  pack_kernel<<<2048, 256, 0, stream>>>(wih_d, whh_d, 1536, KD,   G4,  (_Float16*)(ws + OFF_WD));
  pack_kernel<<<256,  256, 0, stream>>>(wq, wq,       1024, 1024, 128, (_Float16*)(ws + OFF_WQ));
  pack_kernel<<<256,  256, 0, stream>>>(wk, wk,        512, 512,  128, (_Float16*)(ws + OFF_WK));
  pack_kernel<<<256,  256, 0, stream>>>(wm, wm,         KM, KM,   160, (_Float16*)(ws + OFF_WM));
  pack_kernel<<<128,  256, 0, stream>>>(w_pre1, w_pre1, 160, 160, 256, (_Float16*)(ws + OFF_WP1));
  pack_kernel<<<128,  256, 0, stream>>>(w_pre2, w_pre2, 256, 256, 256, (_Float16*)(ws + OFF_WP2));
  init_kernel<<<128, 256, 0, stream>>>(ws, bih_a, bhh_a, bih_d, bhh_d);

  // Prenet over all 16032 rows (WMMA), then memory keys (WMMA).
  prenet1_kernel<<<256, 256, 0, stream>>>(mels, (const _Float16*)(ws + OFF_WP1), b_pre1,
                                          (_Float16*)(ws + OFF_H1));
  prenet2_kernel<<<256, 256, 0, stream>>>((const _Float16*)(ws + OFF_H1),
                                          (const _Float16*)(ws + OFF_WP2), b_pre2,
                                          (_Float16*)(ws + OFF_PRE));
  memkeys_kernel<<<256, 256, 0, stream>>>(enc, (const _Float16*)(ws + OFF_WK), bk,
                                          (float*)(ws + OFF_MEMK));

  // Persistent sequential decoder: 64 blocks x 256 threads = 512 wave32 waves,
  // with 160 KB dynamic LDS per workgroup for activation (A-matrix) staging.
  decoder_kernel<<<64, 256, LDS_BYTES, stream>>>(ws, enc, mask, bq, convw, convb,
                                                 wl, bl, wa, ba, bm, wg, bg,
                                                 out_m, out_g, out_a);
}
